// OptimizedMoELayer_50964081934892
// MI455X (gfx1250) — compile-verified, matrix operand order checked by reading, not run on hardware
//
#include <hip/hip_runtime.h>
#include <math.h>
#include <stdint.h>

// ---------------- problem constants ----------------
#define H       512
#define F       2048
#define E       16
#define NSHARED 2
#define TOPK    4
#define T       4096          // B*S
#define MT      64            // tokens per workgroup tile
#define FC      32            // F-chunk width (double-buffered)
#define NCH     (F / FC)      // 64 chunks
#define TILES   (T / MT)      // 64 token tiles
#define LDSPAD  8
#define HP      (H + LDSPAD)  // 520 halves  (1040 B rows, 16B aligned)
#define FCP     (FC + LDSPAD) // 40 halves   (80 B rows, 16B aligned)

typedef __attribute__((ext_vector_type(16))) _Float16 v16h;
typedef __attribute__((ext_vector_type(8)))  _Float16 v8h;
typedef __attribute__((ext_vector_type(4)))  _Float16 v4h;
typedef __attribute__((ext_vector_type(8)))  float    v8f;
typedef __attribute__((ext_vector_type(4)))  unsigned int u32x4;
typedef __attribute__((ext_vector_type(8)))  int       i32x8;
typedef __attribute__((ext_vector_type(4)))  int       i32x4;

// ---------------- fp32 -> fp16 activation convert ----------------
__global__ void cvt_kernel(const float* __restrict__ x, _Float16* __restrict__ xh, int n4) {
    int i = blockIdx.x * blockDim.x + threadIdx.x;
    if (i < n4) {
        float4 v = ((const float4*)x)[i];
        v4h o = { (_Float16)v.x, (_Float16)v.y, (_Float16)v.z, (_Float16)v.w };
        ((v4h*)xh)[i] = o;
    }
}

// ------------- tiled transpose + fp32->fp16 convert: in[b][R][C] -> out[b][C][R]
__global__ void tconv_kernel(const float* __restrict__ in, _Float16* __restrict__ outp,
                             int R, int C) {
    __shared__ float tile[32][33];
    int b  = blockIdx.z;
    int r0 = blockIdx.y * 32, c0 = blockIdx.x * 32;
    const float* src = in + (size_t)b * R * C;
    _Float16* dst = outp + (size_t)b * R * C;
    int tx = threadIdx.x & 31, ty = threadIdx.x >> 5;   // 256 thr = 32x8
#pragma unroll
    for (int i = 0; i < 32; i += 8)
        tile[ty + i][tx] = src[(size_t)(r0 + ty + i) * C + c0 + tx];
    __syncthreads();
#pragma unroll
    for (int i = 0; i < 32; i += 8)
        dst[(size_t)(c0 + ty + i) * R + r0 + tx] = (_Float16)tile[tx][ty + i];
}

// ---------------- gating: softmax -> top4 -> softmax(top vals) ----------------
__global__ void gate_kernel(const float* __restrict__ x, const float* __restrict__ gw,
                            int* __restrict__ counts, int* __restrict__ list,
                            float* __restrict__ wlist) {
    int wave = threadIdx.x >> 5;
    int lane = threadIdx.x & 31;
    int t = blockIdx.x * 8 + wave;
    if (t >= T) return;

    float acc[E];
#pragma unroll
    for (int e = 0; e < E; ++e) acc[e] = 0.0f;
    for (int h = lane; h < H; h += 32) {
        float xv = x[(size_t)t * H + h];
#pragma unroll
        for (int e = 0; e < E; ++e) acc[e] += xv * gw[(size_t)e * H + h];
    }
#pragma unroll
    for (int e = 0; e < E; ++e) {
#pragma unroll
        for (int off = 16; off > 0; off >>= 1)
            acc[e] += __shfl_xor(acc[e], off, 32);
    }
    if (lane == 0) {
        float m = acc[0];
#pragma unroll
        for (int e = 1; e < E; ++e) m = fmaxf(m, acc[e]);
        float p[E], s = 0.0f;
#pragma unroll
        for (int e = 0; e < E; ++e) { p[e] = __expf(acc[e] - m); s += p[e]; }
        float inv = 1.0f / s;
#pragma unroll
        for (int e = 0; e < E; ++e) p[e] *= inv;

        int idx[TOPK]; float val[TOPK]; bool taken[E];
#pragma unroll
        for (int e = 0; e < E; ++e) taken[e] = false;
#pragma unroll
        for (int k = 0; k < TOPK; ++k) {
            int best = 0; float bv = -1.0f;
#pragma unroll
            for (int e = 0; e < E; ++e)
                if (!taken[e] && p[e] > bv) { bv = p[e]; best = e; }
            taken[best] = true; idx[k] = best; val[k] = bv;
        }
        float m2 = val[0], s2 = 0.0f, w[TOPK];
#pragma unroll
        for (int k = 0; k < TOPK; ++k) { w[k] = __expf(val[k] - m2); s2 += w[k]; }
        float inv2 = 1.0f / s2;
#pragma unroll
        for (int k = 0; k < TOPK; ++k) {
            int e = idx[k];
            int pos = atomicAdd(&counts[e], 1);
            list[e * T + pos]  = t;
            wlist[e * T + pos] = w[k] * inv2;
        }
    }
}

// ---------------- WMMA 16-bit A/B fragment loader ----------------
// lanes 0-15: K in {0..7} U {16..23}; lanes 16-31: K in {8..15} U {24..31}
__device__ inline v16h load_frag(const _Float16* base, int stride, int row, int kBase, int half) {
    const _Float16* p = base + row * stride + kBase + 8 * half;
    v8h lo = *(const v8h*)p;
    v8h hi = *(const v8h*)(p + 16);
    v16h r;
#pragma unroll
    for (int i = 0; i < 8; ++i) { r[i] = lo[i]; r[i + 8] = hi[i]; }
    return r;
}

// ---------------- TDM: one 2-D tile global->LDS, with LDS row padding -------
// flags: data_size=2B, pad_enable, pad_interval (8B units exp), pad_amount (dwords-1)
__device__ inline void tdm_issue(const _Float16* gbase, unsigned ldsOff, unsigned flags,
                                 unsigned tdim0, unsigned tdim1,
                                 unsigned tile0, unsigned tile1, unsigned stride0) {
    unsigned long long ga = (unsigned long long)(uintptr_t)gbase;
    u32x4 g0 = { 1u,                                   // count=1, user descriptor
                 ldsOff,                               // lds_addr (bytes)
                 (unsigned)ga,                         // global_addr[31:0]
                 ((unsigned)(ga >> 32) & 0x1FFFFFFu) | 0x80000000u }; // addr[56:32] | type=2
    i32x8 g1 = { (int)flags,
                 (int)((tdim0 & 0xFFFFu) << 16),                       // tensor_dim0 lo
                 (int)((tdim0 >> 16) | ((tdim1 & 0xFFFFu) << 16)),     // dim0 hi | dim1 lo
                 (int)((tdim1 >> 16) | (tile0 << 16)),                 // dim1 hi | tile_dim0
                 (int)tile1,                                           // tile_dim1 (tile2=0)
                 (int)stride0,                                         // dim0_stride lo
                 0, 0 };
    i32x4 z4 = { 0, 0, 0, 0 };
#if defined(__clang_major__) && (__clang_major__ >= 23)
    i32x8 z8 = { 0, 0, 0, 0, 0, 0, 0, 0 };
    __builtin_amdgcn_tensor_load_to_lds(g0, g1, z4, z4, z8, 0);
#else
    __builtin_amdgcn_tensor_load_to_lds(g0, g1, z4, z4, 0);
#endif
}

// flags: data_size=1 (2 bytes) at bit16, pad_enable bit20, interval<<22, amount<<25
#define TDM_FLAGS_W1 (0x00010000u | 0x00100000u | (7u << 22) | (3u << 25)) // pad every 1024B, +16B
#define TDM_FLAGS_W2 (0x00010000u | 0x00100000u | (3u << 22) | (3u << 25)) // pad every 64B,  +16B

// ---------------- fused expert MLP, TDM double-buffered ----------------
// grid.x = numExperts * TILES ; 256 threads = 8 wave32
__global__ __launch_bounds__(256)
void moe_mlp_kernel(const _Float16* __restrict__ Xh,
                    const _Float16* __restrict__ W1t,  // [e][F][H] f16 (pre-transposed)
                    const float*    __restrict__ B1,
                    const _Float16* __restrict__ W2t,  // [e][H][F] f16 (pre-transposed)
                    const float*    __restrict__ B2,
                    const int* __restrict__ counts, const int* __restrict__ list,
                    const float* __restrict__ wlist, float* __restrict__ out,
                    int sharedMode) {
    extern __shared__ char smem[];
    _Float16* Xs   = (_Float16*)smem;                 // [MT][H]        64 KB
    _Float16* W1b0 = Xs + MT * H;                     // [FC][HP]       32.5 KB
    _Float16* W1b1 = W1b0 + FC * HP;
    _Float16* W2b0 = W1b1 + FC * HP;                  // [H][FCP]       40 KB
    _Float16* W2b1 = W2b0 + H * FCP;
    _Float16* Hs   = W2b1 + H * FCP;                  // [MT][FCP]      5 KB
    int*   tokS = (int*)(Hs + MT * FCP);
    float* wS   = (float*)(tokS + MT);

    const int tid  = threadIdx.x;
    const int e    = blockIdx.x / TILES;
    const int tile = blockIdx.x % TILES;
    const int base = tile * MT;

    int cnt = T;
    if (!sharedMode) {
        cnt = counts[e];
        if (base >= cnt) return;                      // uniform early exit
    }

    const int wv   = tid >> 5;
    const int lane = tid & 31;
    const int half = lane >> 4;
    const int l15  = lane & 15;

    const _Float16* W1e = W1t + (size_t)e * F * H;
    const _Float16* W2e = W2t + (size_t)e * H * F;
    const unsigned w1Off[2] = { (unsigned)(uintptr_t)W1b0, (unsigned)(uintptr_t)W1b1 };
    const unsigned w2Off[2] = { (unsigned)(uintptr_t)W2b0, (unsigned)(uintptr_t)W2b1 };

    // ---- prologue: TDM chunk 0 into buffer 0, overlapped with gather/X staging
    if (wv == 0) {
        tdm_issue(W1e, w1Off[0], TDM_FLAGS_W1, H, FC, H, FC, H);          // FC rows of 512 f16
        tdm_issue(W2e, w2Off[0], TDM_FLAGS_W2, F, H, FC, H, F);           // 512 rows of 32 f16
    }

    if (tid < MT) {
        int p = base + tid;
        if (sharedMode)      { tokS[tid] = p;               wS[tid] = 1.0f / NSHARED; }
        else if (p < cnt)    { tokS[tid] = list[e * T + p]; wS[tid] = wlist[e * T + p]; }
        else                 { tokS[tid] = 0;               wS[tid] = 0.0f; }
    }
    __syncthreads();

    {   // stage gathered X tile: 4 threads/row, 128 halves each
        int r = tid >> 2, seg = (tid & 3) * 128;
        const v8h* src = (const v8h*)(Xh + (size_t)tokS[r] * H + seg);
        v8h* dst = (v8h*)(Xs + r * H + seg);
#pragma unroll
        for (int i = 0; i < 16; ++i) dst[i] = src[i];
    }

    // wave work mapping
    const int mt1 = wv & 3;                  // GEMM1: 1 tile/wave over 64x32
    const int nt1 = wv >> 2;
    const int mt2 = wv & 3;                  // GEMM2: 16 tiles/wave over 64x512
    const int ntOff = (wv >> 2) * 16;
    const int rowA1 = mt1 * 16 + l15;
    const int rowB1 = nt1 * 16 + l15;
    const int rowA2 = mt2 * 16 + l15;

    v8f zero = {};
    v8f accO[16];
#pragma unroll
    for (int j = 0; j < 16; ++j) accO[j] = zero;

    for (int ci = 0; ci < NCH; ++ci) {
        const int p = ci & 1;
        const int f0 = ci * FC;
        // ---- issue next chunk into the other buffer, wait for current chunk
        if (ci + 1 < NCH) {
            if (wv == 0) {
                const int fn = f0 + FC;
                tdm_issue(W1e + (size_t)fn * H, w1Off[1 - p], TDM_FLAGS_W1, H, FC, H, FC, H);
                tdm_issue(W2e + fn,             w2Off[1 - p], TDM_FLAGS_W2, F, H, FC, H, F);
            }
            __builtin_amdgcn_s_wait_tensorcnt(2);
        } else {
            __builtin_amdgcn_s_wait_tensorcnt(0);
        }
        __syncthreads();

        const _Float16* W1p = p ? W1b1 : W1b0;
        const _Float16* W2p = p ? W2b1 : W2b0;

        // ---- GEMM1: Hs[64,32] = gelu(Xs[64,512] @ W1p^T + b1), K pipelined
        v8f c0 = zero;
        v16h a0 = load_frag(Xs,  H,  rowA1, 0, half);
        v16h b0 = load_frag(W1p, HP, rowB1, 0, half);
#pragma unroll
        for (int kb = 0; kb < H - 32; kb += 32) {
            v16h a1 = load_frag(Xs,  H,  rowA1, kb + 32, half);
            v16h b1 = load_frag(W1p, HP, rowB1, kb + 32, half);
            c0 = __builtin_amdgcn_wmma_f32_16x16x32_f16(false, a0, false, b0, (short)0, c0, false, false);
            a0 = a1; b0 = b1;
        }
        c0 = __builtin_amdgcn_wmma_f32_16x16x32_f16(false, a0, false, b0, (short)0, c0, false, false);

        {   // bias + exact GELU -> Hs (C layout: row = v + 8*half, col = lane&15)
            int col = nt1 * 16 + l15;
            float bia = B1[(size_t)e * F + f0 + col];
#pragma unroll
            for (int v = 0; v < 8; ++v) {
                int row = mt1 * 16 + v + 8 * half;
                float xv = c0[v] + bia;
                float g = 0.5f * xv * (1.0f + erff(xv * 0.70710678f));
                Hs[row * FCP + col] = (_Float16)g;
            }
        }
        __syncthreads();

        // ---- GEMM2: accO[64,512] += Hs[64,32] @ W2p^T  (K=32, B double-buffered)
        {
            v16h a = load_frag(Hs, FCP, rowA2, 0, half);
            v16h bcur = load_frag(W2p, FCP, ntOff * 16 + l15, 0, half);
#pragma unroll
            for (int j = 0; j < 15; ++j) {
                v16h bn = load_frag(W2p, FCP, (ntOff + j + 1) * 16 + l15, 0, half);
                accO[j] = __builtin_amdgcn_wmma_f32_16x16x32_f16(false, a, false, bcur, (short)0, accO[j], false, false);
                bcur = bn;
            }
            accO[15] = __builtin_amdgcn_wmma_f32_16x16x32_f16(false, a, false, bcur, (short)0, accO[15], false, false);
        }
        __syncthreads();   // Hs + W-buffer reuse fence
    }

    // ---- epilogue: (acc + b2) * gate weight, scatter-add
#pragma unroll
    for (int j = 0; j < 16; ++j) {
        int hcol = (ntOff + j) * 16 + l15;
        float b2 = B2[(size_t)e * H + hcol];
#pragma unroll
        for (int v = 0; v < 8; ++v) {
            int r = mt2 * 16 + v + 8 * half;
            float val = (accO[j][v] + b2) * wS[r];
            atomicAdd(&out[(size_t)tokS[r] * H + hcol], val);
        }
    }
}

// ---------------- host-side launch ----------------
extern "C" void kernel_launch(void* const* d_in, const int* in_sizes, int n_in,
                              void* d_out, int out_size, void* d_ws, size_t ws_size,
                              hipStream_t stream) {
    const float* x   = (const float*)d_in[0];
    const float* gw  = (const float*)d_in[1];
    const float* rw1 = (const float*)d_in[2];
    const float* rb1 = (const float*)d_in[3];
    const float* rw2 = (const float*)d_in[4];
    const float* rb2 = (const float*)d_in[5];
    const float* sw1 = (const float*)d_in[6];
    const float* sb1 = (const float*)d_in[7];
    const float* sw2 = (const float*)d_in[8];
    const float* sb2 = (const float*)d_in[9];
    float* out = (float*)d_out;

    // workspace layout
    char* ws = (char*)d_ws;
    size_t off = 0;
    _Float16* Xh    = (_Float16*)(ws + off); off += (size_t)T * H * 2;
    int*      cnts  = (int*)(ws + off);      off += 256;
    int*      list  = (int*)(ws + off);      off += (size_t)E * T * 4;
    float*    wlist = (float*)(ws + off);    off += (size_t)E * T * 4;
    _Float16* W1t   = (_Float16*)(ws + off); off += (size_t)E * F * H * 2;       // [e][F][H]
    _Float16* W2t   = (_Float16*)(ws + off); off += (size_t)E * H * F * 2;       // [e][H][F]
    _Float16* S1t   = (_Float16*)(ws + off); off += (size_t)NSHARED * F * H * 2;
    _Float16* S2t   = (_Float16*)(ws + off); off += (size_t)NSHARED * H * F * 2;

    hipMemsetAsync(out, 0, (size_t)T * H * sizeof(float), stream);
    hipMemsetAsync(cnts, 0, E * sizeof(int), stream);

    // activations -> f16
    int n4 = (T * H) / 4;
    cvt_kernel<<<(n4 + 255) / 256, 256, 0, stream>>>(x, Xh, n4);

    // weights -> f16, transposed to B-fragment layout (one-time per call)
    tconv_kernel<<<dim3(F / 32, H / 32, E), 256, 0, stream>>>(rw1, W1t, H, F);       // [H][F]->[F][H]
    tconv_kernel<<<dim3(H / 32, F / 32, E), 256, 0, stream>>>(rw2, W2t, F, H);       // [F][H]->[H][F]
    tconv_kernel<<<dim3(F / 32, H / 32, NSHARED), 256, 0, stream>>>(sw1, S1t, H, F);
    tconv_kernel<<<dim3(H / 32, F / 32, NSHARED), 256, 0, stream>>>(sw2, S2t, F, H);

    // gating + routing lists
    gate_kernel<<<T / 8, 256, 0, stream>>>(x, gw, cnts, list, wlist);

    size_t smem = (size_t)MT * H * 2          // Xs
                + 2 * (size_t)FC * HP * 2     // W1b x2
                + 2 * (size_t)H * FCP * 2     // W2b x2
                + (size_t)MT * FCP * 2        // Hs
                + MT * 4 + MT * 4;            // tokS, wS

    moe_mlp_kernel<<<E * TILES, 256, smem, stream>>>(Xh, W1t, rb1, W2t, rb2,
                                                     cnts, list, wlist, out, 0);
    moe_mlp_kernel<<<NSHARED * TILES, 256, smem, stream>>>(Xh, S1t, sb1, S2t, sb2,
                                                           cnts, list, wlist, out, 1);
}
